// MicroBiomeNetV4_85693187490542
// MI455X (gfx1250) — matrix operand order repaired and checked by reading, not compiled, not run on hardware
//
#include <hip/hip_runtime.h>
#include <cmath>

// ---------------------------------------------------------------------------
// CDNA5 (gfx1250) wave32 WMMA types & helpers
// ---------------------------------------------------------------------------
typedef __attribute__((ext_vector_type(16))) __bf16 v16bf;
typedef __attribute__((ext_vector_type(8)))  float  v8f;

union BF16x16 { v16bf v; __bf16 e[16]; };
union F32x8   { v8f   v; float  e[8];  };
union BF8U    { float4 f4; __bf16 b[8]; };

__device__ __forceinline__ v8f wmma_bf16(v16bf a, v16bf b, v8f c) {
  return __builtin_amdgcn_wmma_f32_16x16x32_bf16(false, a, false, b, (short)0, c,
                                                 false, false);
}

__device__ __forceinline__ void zero8(F32x8& c) {
#pragma unroll
  for (int i = 0; i < 8; ++i) c.e[i] = 0.0f;
}

__device__ __forceinline__ float gelu_exact(float x) {
  return 0.5f * x * (1.0f + erff(x * 0.70710678118654752f));
}

// A-fragment (16x32, bf16 source): K elems [h*8..+7] and [16+h*8..+7]
// -> two contiguous 16B groups.
__device__ __forceinline__ v16bf load_a_frag_bf(const __bf16* __restrict__ base) {
  BF8U lo = *(const BF8U*)(base);
  BF8U hi = *(const BF8U*)(base + 16);
  BF16x16 r;
#pragma unroll
  for (int e = 0; e < 8; ++e) { r.e[e] = lo.b[e]; r.e[8 + e] = hi.b[e]; }
  return r.v;
}

// B-fragment (32x16, bf16 source): K elems [h*8..+7] and [16+h*8..+7].
__device__ __forceinline__ v16bf load_b_frag_bf(const __bf16* __restrict__ base) {
  BF8U lo = *(const BF8U*)(base);
  BF8U hi = *(const BF8U*)(base + 16);
  BF16x16 r;
#pragma unroll
  for (int e = 0; e < 8; ++e) { r.e[e] = lo.b[e]; r.e[8 + e] = hi.b[e]; }
  return r.v;
}

// A-fragment from fp32 source (used for softmax P matrix only).
__device__ __forceinline__ v16bf load_a_frag_f32(const float* __restrict__ base) {
  const float4* p = (const float4*)base;
  float4 a0 = p[0], a1 = p[1], a2 = p[4], a3 = p[5];
  BF16x16 r;
  r.e[0]  = (__bf16)a0.x; r.e[1]  = (__bf16)a0.y; r.e[2]  = (__bf16)a0.z; r.e[3]  = (__bf16)a0.w;
  r.e[4]  = (__bf16)a1.x; r.e[5]  = (__bf16)a1.y; r.e[6]  = (__bf16)a1.z; r.e[7]  = (__bf16)a1.w;
  r.e[8]  = (__bf16)a2.x; r.e[9]  = (__bf16)a2.y; r.e[10] = (__bf16)a2.z; r.e[11] = (__bf16)a2.w;
  r.e[12] = (__bf16)a3.x; r.e[13] = (__bf16)a3.y; r.e[14] = (__bf16)a3.z; r.e[15] = (__bf16)a3.w;
  return r.v;
}

// ---------------------------------------------------------------------------
// WMMA GEMM: C[M,N] = act(A[M,K]bf16 @ W[N,K]bf16^T + bias) (+ residual R f32)
// M%16==0, N%32==0, Kt%32==0. Two N-tiles/wave sharing the A fragment.
// act: 0=none, 1=gelu. Output: Cf (f32) or Cb (bf16), one non-null.
// grid=(N/32, ceil(M/128)), block=256.
// ---------------------------------------------------------------------------
__global__ __launch_bounds__(256)
void gemm_bf16(const __bf16* __restrict__ A, const __bf16* __restrict__ W,
               const float* __restrict__ bias, const float* __restrict__ R,
               float* __restrict__ Cf, __bf16* __restrict__ Cb,
               int M, int N, int Kt, int act) {
  const int lane = threadIdx.x & 31;
  const int wv   = threadIdx.x >> 5;
  const int m0   = (blockIdx.y * 8 + wv) * 16;
  if (m0 >= M) return;  // wave-uniform
  const int n0 = blockIdx.x * 32;
  const int h  = lane >> 4;
  const int l4 = lane & 15;

  const __bf16* arow  = A + (size_t)(m0 + l4) * Kt + h * 8;
  const __bf16* wrow0 = W + (size_t)(n0 + l4) * Kt + h * 8;
  const __bf16* wrow1 = wrow0 + (size_t)16 * Kt;

  F32x8 acc0, acc1; zero8(acc0); zero8(acc1);
  for (int k0 = 0; k0 < Kt; k0 += 32) {
    const v16bf a  = load_a_frag_bf(arow + k0);
    const v16bf b0 = load_b_frag_bf(wrow0 + k0);
    const v16bf b1 = load_b_frag_bf(wrow1 + k0);
    acc0.v = wmma_bf16(a, b0, acc0.v);
    acc1.v = wmma_bf16(a, b1, acc1.v);
  }

#pragma unroll
  for (int t = 0; t < 2; ++t) {
    const F32x8& acc = t ? acc1 : acc0;
    const int n = n0 + t * 16 + l4;
    const float bn = bias ? bias[n] : 0.0f;
#pragma unroll
    for (int v = 0; v < 8; ++v) {
      const int m = m0 + h * 8 + v;
      float x = acc.e[v] + bn;
      if (act == 1) x = gelu_exact(x);
      if (R) x += R[(size_t)m * N + n];
      if (Cf) Cf[(size_t)m * N + n] = x;
      else    Cb[(size_t)m * N + n] = (__bf16)x;
    }
  }
}

// ---------------------------------------------------------------------------
// Pad-copy f32->f32 and pad-convert f32->bf16: out[Rout,Cout] <- in[R,Cin].
// ---------------------------------------------------------------------------
__global__ __launch_bounds__(256)
void padcopy_kernel(const float* __restrict__ in, float* __restrict__ out,
                    int R, int Cin, int Rout, int Cout) {
  const int i = blockIdx.x * 256 + threadIdx.x;
  if (i >= Rout * Cout) return;
  const int r = i / Cout, c = i % Cout;
  out[i] = (r < R && c < Cin) ? in[(size_t)r * Cin + c] : 0.0f;
}

__global__ __launch_bounds__(256)
void padcvt_kernel(const float* __restrict__ in, __bf16* __restrict__ out,
                   int R, int Cin, int Rout, int Cout) {
  const int i = blockIdx.x * 256 + threadIdx.x;
  if (i >= Rout * Cout) return;
  const int r = i / Cout, c = i % Cout;
  out[i] = (__bf16)((r < R && c < Cin) ? in[(size_t)r * Cin + c] : 0.0f);
}

// ---------------------------------------------------------------------------
// Top-K=256 per row of importance [B,5024-strided] (pre-sigmoid; monotone).
// ---------------------------------------------------------------------------
__global__ __launch_bounds__(256)
void topk_kernel(const float* __restrict__ imp, const float* __restrict__ clr,
                 int* __restrict__ idx, float* __restrict__ clrg) {
  __shared__ float vals[5000];
  __shared__ float rv[256];
  __shared__ int   ri[256];
  const int b = blockIdx.x, tid = threadIdx.x;
  for (int j = tid; j < 5000; j += 256) vals[j] = imp[(size_t)b * 5024 + j];
  __syncthreads();
  for (int t = 0; t < 256; ++t) {
    float bv = -1e30f; int bi = 0;
    for (int j = tid; j < 5000; j += 256) {
      float x = vals[j];
      if (x > bv) { bv = x; bi = j; }
    }
    rv[tid] = bv; ri[tid] = bi;
    __syncthreads();
    for (int s = 128; s > 0; s >>= 1) {
      if (tid < s && rv[tid + s] > rv[tid]) { rv[tid] = rv[tid + s]; ri[tid] = ri[tid + s]; }
      __syncthreads();
    }
    if (tid == 0) {
      const int w = ri[0];
      idx[b * 256 + t]  = w;
      clrg[b * 256 + t] = clr[(size_t)b * 5000 + w];
      vals[w] = -1e30f;
    }
    __syncthreads();
  }
}

// tokens[b,k,e] = otu_emb[idx[b,k],e] + clr[b,idx]*v_w[e] + v_b[e]  (fp32 h)
__global__ __launch_bounds__(256)
void tokens_kernel(const int* __restrict__ idx, const float* __restrict__ clrg,
                   const float* __restrict__ emb, const float* __restrict__ vw,
                   const float* __restrict__ vb, float* __restrict__ h) {
  const int bk = blockIdx.x;
  const int e  = threadIdx.x;
  const int id = idx[bk];
  const float cv = clrg[bk];
  h[(size_t)bk * 256 + e] = emb[(size_t)id * 256 + e] + cv * vw[e] + vb[e];
}

// ---------------------------------------------------------------------------
// LayerNorm over last dim 256 (f32 in). One wave/row, 8 rows/block.
// Two variants: f32 out (feeds LN/residual) and bf16 out (feeds GEMM A).
// ---------------------------------------------------------------------------
__device__ __forceinline__ void ln_row_compute(const float* __restrict__ X,
                                               const float* __restrict__ g,
                                               const float* __restrict__ bta,
                                               int row, int lane, float* out8) {
  const float4* x = (const float4*)(X + (size_t)row * 256 + lane * 8);
  float4 v0 = x[0], v1 = x[1];
  float vv[8] = {v0.x, v0.y, v0.z, v0.w, v1.x, v1.y, v1.z, v1.w};
  float s = 0.0f;
#pragma unroll
  for (int i = 0; i < 8; ++i) s += vv[i];
#pragma unroll
  for (int o = 16; o >= 1; o >>= 1) s += __shfl_xor(s, o, 32);
  const float mean = s * (1.0f / 256.0f);
  float var = 0.0f;
#pragma unroll
  for (int i = 0; i < 8; ++i) { float d = vv[i] - mean; var += d * d; }
#pragma unroll
  for (int o = 16; o >= 1; o >>= 1) var += __shfl_xor(var, o, 32);
  const float inv = rsqrtf(var * (1.0f / 256.0f) + 1e-5f);
  const float4* gg = (const float4*)(g + lane * 8);
  const float4* bb = (const float4*)(bta + lane * 8);
  float4 g0 = gg[0], g1 = gg[1], b0 = bb[0], b1 = bb[1];
  float gv[8] = {g0.x, g0.y, g0.z, g0.w, g1.x, g1.y, g1.z, g1.w};
  float bv[8] = {b0.x, b0.y, b0.z, b0.w, b1.x, b1.y, b1.z, b1.w};
#pragma unroll
  for (int i = 0; i < 8; ++i) out8[i] = (vv[i] - mean) * inv * gv[i] + bv[i];
}

__global__ __launch_bounds__(256)
void ln_f32_kernel(const float* __restrict__ X, float* __restrict__ Y,
                   const float* __restrict__ g, const float* __restrict__ bta, int rows) {
  const int lane = threadIdx.x & 31;
  const int row  = blockIdx.x * 8 + (threadIdx.x >> 5);
  if (row >= rows) return;
  float o[8];
  ln_row_compute(X, g, bta, row, lane, o);
  float4* y = (float4*)(Y + (size_t)row * 256 + lane * 8);
  y[0] = make_float4(o[0], o[1], o[2], o[3]);
  y[1] = make_float4(o[4], o[5], o[6], o[7]);
}

__global__ __launch_bounds__(256)
void ln_bf16_kernel(const float* __restrict__ X, __bf16* __restrict__ Y,
                    const float* __restrict__ g, const float* __restrict__ bta, int rows) {
  const int lane = threadIdx.x & 31;
  const int row  = blockIdx.x * 8 + (threadIdx.x >> 5);
  if (row >= rows) return;
  float o[8];
  ln_row_compute(X, g, bta, row, lane, o);
  BF8U ob;
#pragma unroll
  for (int i = 0; i < 8; ++i) ob.b[i] = (__bf16)o[i];
  *(float4*)(Y + (size_t)row * 256 + lane * 8) = ob.f4;
}

// ---------------------------------------------------------------------------
// Fused MHA for one (batch, head): seq=256, dh=32, bf16 in (qkv), bf16 out.
// K staged [256,32] bf16, V staged transposed [32,256] bf16, per-wave f32
// score strip. Dynamic LDS = 160 KB.
// ---------------------------------------------------------------------------
__global__ __launch_bounds__(256)
void attn_kernel(const __bf16* __restrict__ qkv, __bf16* __restrict__ out) {
  extern __shared__ float smem[];
  __bf16* Kp = (__bf16*)smem;        // [256][32] bf16 (16 KB)
  __bf16* Vt = Kp + 256 * 32;        // [32][256] bf16 (16 KB)
  float*  Sm = (float*)(Vt + 32 * 256);  // 8 waves * [16][256] f32 (128 KB)
  const int b    = blockIdx.x >> 3;
  const int head = blockIdx.x & 7;
  const int tid  = threadIdx.x;
  const int lane = tid & 31;
  const int wv   = tid >> 5;
  const int h    = lane >> 4;
  const int l4   = lane & 15;
  const __bf16* base = qkv + (size_t)b * 256 * 768;

  for (int i = tid; i < 1024; i += 256) {  // 8-elem groups cover K and V
    const int j = i >> 2, d8 = (i & 3) * 8;
    const __bf16* src = base + (size_t)j * 768 + head * 32 + d8;
    BF8U kf; kf.f4 = *(const float4*)(src + 256);
    BF8U vf; vf.f4 = *(const float4*)(src + 512);
    *(float4*)(Kp + j * 32 + d8) = kf.f4;
#pragma unroll
    for (int u = 0; u < 8; ++u) Vt[(d8 + u) * 256 + j] = vf.b[u];
  }
  __syncthreads();

  float* S = Sm + wv * 16 * 256;
  const float scale = 0.17677669529663687f;  // 1/sqrt(32)

  for (int qt = wv; qt < 16; qt += 8) {
    const int m0 = qt * 16;
    const v16bf a = load_a_frag_bf(base + (size_t)(m0 + l4) * 768 + head * 32 + h * 8);
    // S = scale * Q @ K^T
    for (int nt = 0; nt < 16; ++nt) {
      const v16bf bm = load_b_frag_bf(Kp + (nt * 16 + l4) * 32 + h * 8);
      F32x8 c; zero8(c);
      c.v = wmma_bf16(a, bm, c.v);
#pragma unroll
      for (int v = 0; v < 8; ++v)
        S[(h * 8 + v) * 256 + nt * 16 + l4] = c.e[v] * scale;
    }
    // wave-local softmax per query row
    for (int r = 0; r < 16; ++r) {
      float* row = S + r * 256;
      float mx = -1e30f;
#pragma unroll
      for (int i = 0; i < 8; ++i) mx = fmaxf(mx, row[lane + 32 * i]);
#pragma unroll
      for (int o = 16; o >= 1; o >>= 1) mx = fmaxf(mx, __shfl_xor(mx, o, 32));
      float ex[8]; float sum = 0.0f;
#pragma unroll
      for (int i = 0; i < 8; ++i) { ex[i] = __expf(row[lane + 32 * i] - mx); sum += ex[i]; }
#pragma unroll
      for (int o = 16; o >= 1; o >>= 1) sum += __shfl_xor(sum, o, 32);
      const float inv = 1.0f / sum;
#pragma unroll
      for (int i = 0; i < 8; ++i) row[lane + 32 * i] = ex[i] * inv;
    }
    // O = P @ V : N=32 (two tiles), K=256 (8 steps)
#pragma unroll
    for (int nt2 = 0; nt2 < 2; ++nt2) {
      const int n0 = nt2 * 16;
      F32x8 c; zero8(c);
      for (int ks = 0; ks < 8; ++ks) {
        const int k0 = ks * 32;
        const v16bf ap = load_a_frag_f32(S + l4 * 256 + k0 + h * 8);
        const v16bf bp = load_b_frag_bf(Vt + (n0 + l4) * 256 + k0 + h * 8);
        c.v = wmma_bf16(ap, bp, c.v);
      }
      const int n = n0 + l4;
#pragma unroll
      for (int v = 0; v < 8; ++v) {
        const int m = m0 + h * 8 + v;
        out[((size_t)b * 256 + m) * 256 + head * 32 + n] = (__bf16)c.e[v];
      }
    }
  }
}

// ---------------------------------------------------------------------------
// Fused FFN per 16-row strip (all bf16 operands): f1 = gelu(hn@W1^T+b1) in
// LDS (bf16); h(f32) += f1@W2^T + b2.
// ---------------------------------------------------------------------------
__global__ __launch_bounds__(256)
void ffn_kernel(const __bf16* __restrict__ hn, const __bf16* __restrict__ w1,
                const float* __restrict__ b1, const __bf16* __restrict__ w2,
                const float* __restrict__ b2, float* __restrict__ h) {
  __shared__ __bf16 xs[16 * 256];     // 8 KB input strip
  __shared__ __bf16 f1[16 * 1024];    // 32 KB activated hidden strip
  const int tid = threadIdx.x;
  const int lane = tid & 31, wv = tid >> 5;
  const int hh = lane >> 4, l4 = lane & 15;
  const size_t row0 = (size_t)blockIdx.x * 16;

  for (int i = tid; i < 512; i += 256)
    ((float4*)xs)[i] = ((const float4*)(hn + row0 * 256))[i];
  __syncthreads();

  // phase 1: N=1024 (64 tiles / 8 waves), K=256
  for (int t = 0; t < 8; ++t) {
    const int n0 = (wv + 8 * t) * 16;
    const __bf16* wrow = w1 + (size_t)(n0 + l4) * 256 + hh * 8;
    F32x8 c; zero8(c);
    for (int ks = 0; ks < 8; ++ks) {
      const int k0 = ks * 32;
      const v16bf a  = load_a_frag_bf(xs + l4 * 256 + k0 + hh * 8);
      const v16bf bm = load_b_frag_bf(wrow + k0);
      c.v = wmma_bf16(a, bm, c.v);
    }
    const int n = n0 + l4;
    const float bb = b1[n];
#pragma unroll
    for (int v = 0; v < 8; ++v)
      f1[(hh * 8 + v) * 1024 + n] = (__bf16)gelu_exact(c.e[v] + bb);
  }
  __syncthreads();

  // phase 2: N=256 (16 tiles / 8 waves), K=1024
#pragma unroll
  for (int t = 0; t < 2; ++t) {
    const int n0 = (wv + 8 * t) * 16;
    const __bf16* wrow = w2 + (size_t)(n0 + l4) * 1024 + hh * 8;
    F32x8 c; zero8(c);
    for (int ks = 0; ks < 32; ++ks) {
      const int k0 = ks * 32;
      const v16bf a  = load_a_frag_bf(f1 + l4 * 1024 + k0 + hh * 8);
      const v16bf bm = load_b_frag_bf(wrow + k0);
      c.v = wmma_bf16(a, bm, c.v);
    }
    const int n = n0 + l4;
    const float bb = b2[n];
#pragma unroll
    for (int v = 0; v < 8; ++v) {
      const size_t r = row0 + hh * 8 + v;
      h[r * 256 + n] += c.e[v] + bb;
    }
  }
}

// qp[e2] = pool_q @ in_w[:E]^T + in_b (single query, batch-invariant, f32)
__global__ __launch_bounds__(256)
void qp_kernel(const float* __restrict__ pq, const float* __restrict__ inw,
               const float* __restrict__ inb, float* __restrict__ qp) {
  const int e2 = threadIdx.x;
  float s = inb[e2];
  const float4* w = (const float4*)(inw + e2 * 256);
  const float4* q = (const float4*)pq;
  for (int e = 0; e < 64; ++e) {
    float4 a = q[e], b = w[e];
    s += a.x * b.x + a.y * b.y + a.z * b.z + a.w * b.w;
  }
  qp[e2] = s;
}

// 1-query, 4-head (dh=64) attention pooling per batch row; bf16 pooled out.
__global__ __launch_bounds__(256)
void pool_attn_kernel(const float* __restrict__ qp, const float* __restrict__ kp,
                      const float* __restrict__ vp, __bf16* __restrict__ pooled) {
  __shared__ float sc[4 * 256];
  __shared__ float red[256];
  const int b = blockIdx.x, tid = threadIdx.x;
  const float scale = 0.125f;  // 1/sqrt(64)
  {
    const float* krow = kp + ((size_t)b * 256 + tid) * 256;
    for (int hh = 0; hh < 4; ++hh) {
      const float4* q = (const float4*)(qp + hh * 64);
      const float4* k = (const float4*)(krow + hh * 64);
      float s = 0.0f;
      for (int d = 0; d < 16; ++d) {
        float4 a = q[d], c = k[d];
        s += a.x * c.x + a.y * c.y + a.z * c.z + a.w * c.w;
      }
      sc[hh * 256 + tid] = s * scale;
    }
  }
  __syncthreads();
  for (int hh = 0; hh < 4; ++hh) {
    red[tid] = sc[hh * 256 + tid];
    __syncthreads();
    for (int s = 128; s > 0; s >>= 1) {
      if (tid < s) red[tid] = fmaxf(red[tid], red[tid + s]);
      __syncthreads();
    }
    const float mx = red[0];
    __syncthreads();
    const float e = __expf(sc[hh * 256 + tid] - mx);
    red[tid] = e;
    __syncthreads();
    for (int s = 128; s > 0; s >>= 1) {
      if (tid < s) red[tid] += red[tid + s];
      __syncthreads();
    }
    const float sm = red[0];
    __syncthreads();
    sc[hh * 256 + tid] = e / sm;
  }
  __syncthreads();
  const int hh = tid >> 6, d = tid & 63;
  float acc = 0.0f;
  for (int j = 0; j < 256; ++j)
    acc += sc[hh * 256 + j] * vp[((size_t)b * 256 + j) * 256 + hh * 64 + d];
  pooled[(size_t)b * 256 + tid] = (__bf16)acc;
}

// logits (stride 32, first 8 valid) + softmax -> d_out
__global__ __launch_bounds__(128)
void final_kernel(const float* __restrict__ logits, float* __restrict__ out) {
  const int r = threadIdx.x;
  float v[8]; float mx = -1e30f;
#pragma unroll
  for (int i = 0; i < 8; ++i) { v[i] = logits[r * 32 + i]; mx = fmaxf(mx, v[i]); }
  float ex[8]; float s = 0.0f;
#pragma unroll
  for (int i = 0; i < 8; ++i) { ex[i] = __expf(v[i] - mx); s += ex[i]; }
  const float inv = 1.0f / s;
#pragma unroll
  for (int i = 0; i < 8; ++i) {
    out[r * 8 + i]        = v[i];
    out[1024 + r * 8 + i] = ex[i] * inv;
  }
}

// ---------------------------------------------------------------------------
// Host orchestration
// ---------------------------------------------------------------------------
extern "C" void kernel_launch(void* const* d_in, const int* in_sizes, int n_in,
                              void* d_out, int out_size, void* d_ws, size_t ws_size,
                              hipStream_t stream) {
  (void)in_sizes; (void)n_in; (void)out_size; (void)ws_size;
  constexpr int B = 128, D = 5000, Dp = 5024, E = 256, H = 8, L = 6, FF = 1024, K = 256;
  constexpr int MT = B * K;  // 32768 token rows

  // Inputs in setup_inputs() dict (insertion) order:
  const float* clr       = (const float*)d_in[0];
  const float* g_w1      = (const float*)d_in[1];
  const float* g_b1      = (const float*)d_in[2];
  const float* g_w2      = (const float*)d_in[3];
  const float* g_b2      = (const float*)d_in[4];
  const float* otu_emb   = (const float*)d_in[5];
  const float* v_w       = (const float*)d_in[6];
  const float* v_b       = (const float*)d_in[7];
  const float* in_w      = (const float*)d_in[8];    // [L,768,256]
  const float* in_b      = (const float*)d_in[9];
  const float* out_w     = (const float*)d_in[10];   // [L,256,256]
  const float* out_b     = (const float*)d_in[11];
  const float* ln1_g     = (const float*)d_in[12];
  const float* ln1_b     = (const float*)d_in[13];
  const float* ln2_g     = (const float*)d_in[14];
  const float* ln2_b     = (const float*)d_in[15];
  const float* ff_w1     = (const float*)d_in[16];   // [L,1024,256]
  const float* ff_b1     = (const float*)d_in[17];
  const float* ff_w2     = (const float*)d_in[18];   // [L,256,1024]
  const float* ff_b2     = (const float*)d_in[19];
  const float* pool_q    = (const float*)d_in[20];
  const float* pool_in_w = (const float*)d_in[21];   // [768,256]
  const float* pool_in_b = (const float*)d_in[22];
  const float* pool_out_w= (const float*)d_in[23];
  const float* pool_out_b= (const float*)d_in[24];
  const float* pool_ln_g = (const float*)d_in[25];
  const float* pool_ln_b = (const float*)d_in[26];
  const float* c_ln_g    = (const float*)d_in[27];
  const float* c_ln_b    = (const float*)d_in[28];
  const float* c_w1      = (const float*)d_in[29];
  const float* c_b1      = (const float*)d_in[30];
  const float* c_w2      = (const float*)d_in[31];
  const float* c_b2      = (const float*)d_in[32];
  const float* c_w3      = (const float*)d_in[33];
  const float* c_b3      = (const float*)d_in[34];

  // ---- Workspace carve: f32 region, then bf16 region ----
  float* wf = (float*)d_ws;
  float* ws_h    = wf; wf += (size_t)MT * E;   // residual stream (f32)
  float* ws_kp   = wf; wf += (size_t)MT * E;   // pooling k-proj (f32)
  float* ws_vp   = wf; wf += (size_t)MT * E;   // pooling v-proj (f32)
  float* ws_imp  = wf; wf += (size_t)B * Dp;
  float* ws_clrg = wf; wf += (size_t)B * K;
  float* ws_qp   = wf; wf += 256;
  float* ws_pool2= wf; wf += (size_t)B * E;
  float* ws_x0   = wf; wf += (size_t)B * E;
  float* ws_lg   = wf; wf += (size_t)B * 32;
  float* ws_gb2p = wf; wf += Dp;               // g_b2 padded (f32)
  float* ws_cb3p = wf; wf += 32;               // c_b3 padded (f32)
  int*   ws_idx  = (int*)wf; wf += (size_t)B * K;

  __bf16* wb = (__bf16*)wf;
  __bf16* bh_hn   = wb; wb += (size_t)MT * E;        // LN outputs
  __bf16* bh_qkv  = wb; wb += (size_t)MT * 3 * E;
  __bf16* bh_ao   = wb; wb += (size_t)MT * E;        // attention out
  __bf16* bh_h    = wb; wb += (size_t)MT * E;        // final h (bf16 copy)
  __bf16* bh_s    = wb; wb += (size_t)B * 512;
  __bf16* bh_clrp = wb; wb += (size_t)B * Dp;
  __bf16* bh_x1   = wb; wb += (size_t)B * E;
  __bf16* bh_x2   = wb; wb += (size_t)B * 512;
  __bf16* bh_x3   = wb; wb += (size_t)B * E;
  __bf16* bh_pool = wb; wb += (size_t)B * E;
  __bf16* w_in    = wb; wb += (size_t)L * 3 * E * E;
  __bf16* w_out   = wb; wb += (size_t)L * E * E;
  __bf16* w_ff1   = wb; wb += (size_t)L * FF * E;
  __bf16* w_ff2   = wb; wb += (size_t)L * E * FF;
  __bf16* w_gw1   = wb; wb += (size_t)512 * Dp;
  __bf16* w_gw2   = wb; wb += (size_t)Dp * 512;
  __bf16* w_pk    = wb; wb += (size_t)E * E;
  __bf16* w_pv    = wb; wb += (size_t)E * E;
  __bf16* w_po    = wb; wb += (size_t)E * E;
  __bf16* w_c1    = wb; wb += (size_t)512 * E;
  __bf16* w_c2    = wb; wb += (size_t)E * 512;
  __bf16* w_c3    = wb; wb += (size_t)32 * E;

  auto gemm_f = [&](const __bf16* A, const __bf16* W, const float* bias, const float* R,
                    float* C, int M, int N, int Kt, int act) {
    dim3 grid(N / 32, (M + 127) / 128);
    gemm_bf16<<<grid, 256, 0, stream>>>(A, W, bias, R, C, nullptr, M, N, Kt, act);
  };
  auto gemm_b = [&](const __bf16* A, const __bf16* W, const float* bias,
                    __bf16* C, int M, int N, int Kt, int act) {
    dim3 grid(N / 32, (M + 127) / 128);
    gemm_bf16<<<grid, 256, 0, stream>>>(A, W, bias, nullptr, nullptr, C, M, N, Kt, act);
  };
  auto cvt = [&](const float* in, __bf16* out, int R, int Cin, int Rout, int Cout) {
    const int total = Rout * Cout;
    padcvt_kernel<<<(total + 255) / 256, 256, 0, stream>>>(in, out, R, Cin, Rout, Cout);
  };
  auto padf = [&](const float* in, float* out, int R, int Cin, int Rout, int Cout) {
    const int total = Rout * Cout;
    padcopy_kernel<<<(total + 255) / 256, 256, 0, stream>>>(in, out, R, Cin, Rout, Cout);
  };

  const int attn_lds = (256 * 32 + 32 * 256) * 2 + 8 * 16 * 256 * 4;  // 160 KB
  (void)hipFuncSetAttribute((const void*)attn_kernel,
                            hipFuncAttributeMaxDynamicSharedMemorySize, attn_lds);

  // ---- One-time (per launch) weight conversion to bf16 (+padding) ----
  cvt(clr,  bh_clrp, B,   D,   B,   Dp);
  cvt(g_w1, w_gw1, 512, D,   512, Dp);
  cvt(g_w2, w_gw2, D,   512, Dp,  512);
  padf(g_b2, ws_gb2p, 1, D, 1, Dp);
  cvt(in_w,  w_in,  L * 3 * E, E, L * 3 * E, E);
  cvt(out_w, w_out, L * E, E, L * E, E);
  cvt(ff_w1, w_ff1, L * FF, E, L * FF, E);
  cvt(ff_w2, w_ff2, L * E, FF, L * E, FF);
  cvt(pool_in_w + E * E,     w_pk, E, E, E, E);
  cvt(pool_in_w + 2 * E * E, w_pv, E, E, E, E);
  cvt(pool_out_w, w_po, E, E, E, E);
  cvt(c_w1, w_c1, 512, E, 512, E);
  cvt(c_w2, w_c2, E, 512, E, 512);
  cvt(c_w3, w_c3, 8, E, 32, E);
  padf(c_b3, ws_cb3p, 1, 8, 1, 32);

  // ---- Gate MLP + top-k selection ----
  gemm_b(bh_clrp, w_gw1, g_b1, bh_s, B, 512, Dp, /*gelu*/1);
  gemm_f(bh_s, w_gw2, ws_gb2p, nullptr, ws_imp, B, Dp, 512, 0);  // sigmoid monotone
  topk_kernel<<<B, 256, 0, stream>>>(ws_imp, clr, ws_idx, ws_clrg);
  tokens_kernel<<<B * K, 256, 0, stream>>>(ws_idx, ws_clrg, otu_emb, v_w, v_b, ws_h);

  // ---- Transformer encoder ----
  for (int i = 0; i < L; ++i) {
    ln_bf16_kernel<<<MT / 8, 256, 0, stream>>>(ws_h, bh_hn, ln1_g + i * E, ln1_b + i * E, MT);
    gemm_b(bh_hn, w_in + (size_t)i * 3 * E * E, in_b + (size_t)i * 3 * E,
           bh_qkv, MT, 3 * E, E, 0);
    attn_kernel<<<B * H, 256, attn_lds, stream>>>(bh_qkv, bh_ao);
    gemm_f(bh_ao, w_out + (size_t)i * E * E, out_b + (size_t)i * E,
           /*residual*/ws_h, ws_h, MT, E, E, 0);
    ln_bf16_kernel<<<MT / 8, 256, 0, stream>>>(ws_h, bh_hn, ln2_g + i * E, ln2_b + i * E, MT);
    ffn_kernel<<<MT / 16, 256, 0, stream>>>(bh_hn, w_ff1 + (size_t)i * FF * E,
                                            ff_b1 + (size_t)i * FF,
                                            w_ff2 + (size_t)i * E * FF,
                                            ff_b2 + (size_t)i * E, ws_h);
  }

  // ---- Attention pooling ----
  cvt(ws_h, bh_h, MT, E, MT, E);  // final h -> bf16 for K/V projections
  qp_kernel<<<1, 256, 0, stream>>>(pool_q, pool_in_w, pool_in_b, ws_qp);
  gemm_f(bh_h, w_pk, pool_in_b + E,     nullptr, ws_kp, MT, E, E, 0);
  gemm_f(bh_h, w_pv, pool_in_b + 2 * E, nullptr, ws_vp, MT, E, E, 0);
  pool_attn_kernel<<<B, 256, 0, stream>>>(ws_qp, ws_kp, ws_vp, bh_pool);
  gemm_f(bh_pool, w_po, pool_out_b, nullptr, ws_pool2, B, E, E, 0);
  ln_f32_kernel<<<(B + 7) / 8, 256, 0, stream>>>(ws_pool2, ws_x0, pool_ln_g, pool_ln_b, B);

  // ---- Classifier head ----
  ln_bf16_kernel<<<(B + 7) / 8, 256, 0, stream>>>(ws_x0, bh_x1, c_ln_g, c_ln_b, B);
  gemm_b(bh_x1, w_c1, c_b1, bh_x2, B, 512, E, 1);
  gemm_b(bh_x2, w_c2, c_b2, bh_x3, B, 256, 512, 1);
  gemm_f(bh_x3, w_c3, ws_cb3p, nullptr, ws_lg, B, 32, 256, 0);
  final_kernel<<<1, 128, 0, stream>>>(ws_lg, (float*)d_out);
}
